// LDPC5GEncoder_60653528154575
// MI455X (gfx1250) — 8-bit matrix operand order repaired and checked by reading, compiled, not run on hardware
//
#include <hip/hip_runtime.h>
#include <hip/hip_bf16.h>

#define K_BITS 8448
#define N_OUT  25344
#define Z_LIFT 384
#define M1_P   1536
#define M2_P   16128
#define B_BATCH 1024
#define DEG_A  19
#define DEG_C1 6
#define DEG_C2 4

typedef __attribute__((ext_vector_type(8))) int v8i;

__global__ __launch_bounds__(256) void pcm_to_u8(const float* __restrict__ pcm,
                                                 unsigned char* __restrict__ w8) {
    int t = blockIdx.x * 256 + threadIdx.x;
    w8[t] = (pcm[t] > 0.5f) ? (unsigned char)1 : (unsigned char)0;
}

__global__ __launch_bounds__(256) void compute_pa(const float* __restrict__ bits,
                                                  const int* __restrict__ ind_a,
                                                  unsigned char* __restrict__ pa8) {
    int t = blockIdx.x * 256 + threadIdx.x;
    int m = t % M1_P;
    int b = t / M1_P;
    const int*   ia  = ind_a + m * DEG_A;
    const float* row = bits + (size_t)b * K_BITS;
    int s = 0;
#pragma unroll
    for (int d = 0; d < DEG_A; ++d) {
        int idx = ia[d];
        s += (idx < K_BITS) ? (int)row[idx] : 0;
    }
    pa8[t] = (unsigned char)(s & 1);
}

__global__ __launch_bounds__(256) void gemm_mod2_wmma(const unsigned char* __restrict__ pa8,
                                                      const unsigned char* __restrict__ w8,
                                                      unsigned char* __restrict__ p18) {
    const int lane  = threadIdx.x & 31;
    const int wave  = threadIdx.x >> 5;
    const int tile  = blockIdx.x * 8 + wave;
    const int JT    = M1_P / 16;
    const int tb    = tile / JT;
    const int tj    = tile % JT;
    const int ridx  = lane & 15;
    const int khalf = (lane >> 4) & 1;

    const unsigned char* Arow = pa8 + (size_t)(tb * 16 + ridx) * M1_P;
    const unsigned char* Brow = w8  + (size_t)(tj * 16 + ridx) * M1_P;

    v8i acc = {0, 0, 0, 0, 0, 0, 0, 0};
#pragma unroll 4
    for (int kk = 0; kk < M1_P / 64; ++kk) {
        const int ka = kk * 64 + khalf * 16;
        uint2 a01 = *(const uint2*)(Arow + ka);
        uint2 a23 = *(const uint2*)(Arow + ka + 8);
        uint2 a45 = *(const uint2*)(Arow + ka + 32);
        uint2 a67 = *(const uint2*)(Arow + ka + 40);
        const int kb = kk * 64 + khalf * 16;
        uint4 b03 = *(const uint4*)(Brow + kb);
        uint4 b47 = *(const uint4*)(Brow + kb + 32);

        v8i A, Bv;
        A[0] = (int)a01.x; A[1] = (int)a01.y; A[2] = (int)a23.x; A[3] = (int)a23.y;
        A[4] = (int)a45.x; A[5] = (int)a45.y; A[6] = (int)a67.x; A[7] = (int)a67.y;
        Bv[0] = (int)b03.x; Bv[1] = (int)b03.y; Bv[2] = (int)b03.z; Bv[3] = (int)b03.w;
        Bv[4] = (int)b47.x; Bv[5] = (int)b47.y; Bv[6] = (int)b47.z; Bv[7] = (int)b47.w;

        acc = __builtin_amdgcn_wmma_i32_16x16x64_iu8(false, A, false, Bv, acc,
                                                     false, false);
    }

    const int ncol  = lane & 15;
    const int mbase = (lane < 16) ? 0 : 8;
#pragma unroll
    for (int v = 0; v < 8; ++v) {
        int row = tb * 16 + mbase + v;
        p18[(size_t)row * M1_P + tj * 16 + ncol] = (unsigned char)(acc[v] & 1);
    }
}

__global__ __launch_bounds__(256) void compute_p2(const float* __restrict__ bits,
                                                  const int* __restrict__ ind_c1,
                                                  const int* __restrict__ ind_c2,
                                                  const unsigned char* __restrict__ p18,
                                                  unsigned char* __restrict__ p28) {
    int t = blockIdx.x * 256 + threadIdx.x;
    int m = t % M2_P;
    int b = t / M2_P;
    const float*         row   = bits + (size_t)b * K_BITS;
    const unsigned char* p1row = p18 + (size_t)b * M1_P;
    int s = 0;
    const int* i1 = ind_c1 + m * DEG_C1;
#pragma unroll
    for (int d = 0; d < DEG_C1; ++d) {
        int idx = i1[d];
        s += (idx < K_BITS) ? (int)row[idx] : 0;
    }
    const int* i2 = ind_c2 + m * DEG_C2;
#pragma unroll
    for (int d = 0; d < DEG_C2; ++d) {
        int idx = i2[d];
        s += (idx < M1_P) ? (int)p1row[idx] : 0;
    }
    p28[t] = (unsigned char)(s & 1);
}

__global__ __launch_bounds__(256) void assemble_out(const float* __restrict__ bits,
                                                    const unsigned char* __restrict__ p18,
                                                    const unsigned char* __restrict__ p28,
                                                    const int* __restrict__ out_int,
                                                    float* __restrict__ out) {
    int t = blockIdx.x * 256 + threadIdx.x;
    int i = t % N_OUT;
    int b = t / N_OUT;
    int j = out_int[i] + 2 * Z_LIFT;
    float v;
    if (j < K_BITS) {
        v = bits[(size_t)b * K_BITS + j];
    } else if (j < K_BITS + M1_P) {
        v = (float)p18[(size_t)b * M1_P + (j - K_BITS)];
    } else {
        v = (float)p28[(size_t)b * M2_P + (j - K_BITS - M1_P)];
    }
    out[t] = v;
}

extern "C" void kernel_launch(void* const* d_in, const int* in_sizes, int n_in,
                              void* d_out, int out_size, void* d_ws, size_t ws_size,
                              hipStream_t stream) {
    const float* bits    = (const float*)d_in[0];
    const float* pcm     = (const float*)d_in[1];
    const int*   ind_a   = (const int*)d_in[2];
    const int*   ind_c1  = (const int*)d_in[3];
    const int*   ind_c2  = (const int*)d_in[4];
    const int*   out_int = (const int*)d_in[5];
    float*       out     = (float*)d_out;

    unsigned char* ws  = (unsigned char*)d_ws;
    unsigned char* w8  = ws;
    unsigned char* pa8 = ws + 2359296;
    unsigned char* p18 = ws + 3932160;
    unsigned char* p28 = ws + 5505024;

    pcm_to_u8<<<(M1_P * M1_P) / 256, 256, 0, stream>>>(pcm, w8);
    compute_pa<<<(B_BATCH * M1_P) / 256, 256, 0, stream>>>(bits, ind_a, pa8);
    gemm_mod2_wmma<<<((B_BATCH / 16) * (M1_P / 16)) / 8, 256, 0, stream>>>(pa8, w8, p18);
    compute_p2<<<(B_BATCH * M2_P) / 256, 256, 0, stream>>>(bits, ind_c1, ind_c2, p18, p28);
    assemble_out<<<(B_BATCH * N_OUT) / 256, 256, 0, stream>>>(bits, p18, p28, out_int, out);
}
